// ExternalMemory_37967510896684
// MI455X (gfx1250) — compile-verified, matrix operand order checked by reading, not run on hardware
//
#include <hip/hip_runtime.h>
#include <hip/hip_bf16.h>

#define D_MODEL 512
#define NSLOTS 4096
#define BATCH_N 8192

typedef __attribute__((ext_vector_type(16))) __bf16 v16bf;
typedef __attribute__((ext_vector_type(8)))  float  v8f;
typedef __attribute__((ext_vector_type(4)))  unsigned int u32x4;
typedef __attribute__((ext_vector_type(8)))  int i32x8;
typedef __attribute__((ext_vector_type(4)))  int i32x4;

union ABReg { v16bf v; unsigned u[8]; };

// round-to-nearest-even fp32 -> bf16, packed pair into one dword
__device__ __forceinline__ unsigned pack2bf(float lo, float hi) {
  unsigned a = __float_as_uint(lo), b = __float_as_uint(hi);
  a = (a + 0x7FFFu + ((a >> 16) & 1u)) >> 16;
  b = (b + 0x7FFFu + ((b >> 16) & 1u)) >> 16;
  return (b << 16) | (a & 0xFFFFu);
}

__device__ __forceinline__ unsigned short f2bf16(float f) {
  unsigned a = __float_as_uint(f);
  return (unsigned short)((a + 0x7FFFu + ((a >> 16) & 1u)) >> 16);
}

// K index of the low element of bf16 pair v (0..7) for this lane-half (hi),
// matching the 16-bit A/B 16x32 WMMA register layout (ISA 7.12.2).
__device__ __forceinline__ int kpair(int kbase, int v, int hi) {
  return kbase + ((v & 4) << 2) + hi * 8 + ((v & 3) << 1);
}

// load one 16x32 bf16 operand tile (8 dwords) from a dword-indexed row base
__device__ __forceinline__ void load_tile(const unsigned* __restrict__ row,
                                          int kb, int hi, ABReg& t) {
#pragma unroll
  for (int v = 0; v < 8; ++v) t.u[v] = row[kpair(kb, v, hi) >> 1];
}

__device__ __forceinline__ void load_b4(const unsigned* const brow[4],
                                        int kb, int hi, ABReg b[4]) {
#pragma unroll
  for (int tb = 0; tb < 4; ++tb) load_tile(brow[tb], kb, hi, b[tb]);
}

__device__ __forceinline__ void wmma4(const ABReg& a, const ABReg b[4], v8f c[4]) {
#pragma unroll
  for (int tb = 0; tb < 4; ++tb)
    c[tb] = __builtin_amdgcn_wmma_f32_16x16x32_bf16(false, a.v, false, b[tb].v,
                                                    (short)0, c[tb], false, false);
}

// ---------------- streaming fp32 -> bf16 conversion ----------------
__global__ void cvt_bf16_kernel(const float* __restrict__ src,
                                unsigned short* __restrict__ dst) {
  const int i = (blockIdx.x * 256 + threadIdx.x) * 4;
  float4 v = *(const float4*)(src + i);
  unsigned* d = (unsigned*)(dst + i);
  d[0] = pack2bf(v.x, v.y);
  d[1] = pack2bf(v.z, v.w);
}

// ---------------- gate = sigmoid(value . w_gate + b_gate) ----------------
__global__ void gate_kernel(const float* __restrict__ Value,
                            const float* __restrict__ wg,
                            const float* __restrict__ bg,
                            float* __restrict__ Gate) {
  const int lane = threadIdx.x & 31;
  const int wave = threadIdx.x >> 5;
  const int row  = blockIdx.x * 8 + wave;
  const float* vr = Value + (size_t)row * D_MODEL;
  float s = 0.f;
  for (int c = lane; c < D_MODEL; c += 32) s += vr[c] * wg[c];
#pragma unroll
  for (int off = 16; off > 0; off >>= 1) s += __shfl_xor(s, off, 32);
  if (lane == 0) Gate[row] = 1.f / (1.f + expf(-(s + bg[0])));
}

// ---------------- memT[j,n] = bf16(memory[n,j]) ----------------
__global__ void transpose_mem_kernel(const float* __restrict__ Mem,
                                     unsigned short* __restrict__ MemT) {
  const int idx = blockIdx.x * 256 + threadIdx.x;   // over 512*4096 outputs
  const int j = idx >> 12;        // output row (0..511)
  const int n = idx & (NSLOTS - 1);
  MemT[idx] = f2bf16(Mem[(size_t)n * D_MODEL + j]);
}

// ---------------- out_bf16[M,512] = A_bf[M,512] @ W_bf^T + bias ----------
// Block covers 16 rows x full 512 cols; wave w owns 4 N-tiles sharing A regs.
// K unrolled by 2 with ping-pong operand buffers (no register copies).
__global__ void proj_gemm_kernel(const unsigned short* __restrict__ Abf,
                                 const unsigned short* __restrict__ Wbf,
                                 const float* __restrict__ bias,
                                 unsigned short* __restrict__ OutBf) {
  const int lane  = threadIdx.x & 31;
  const int wave  = threadIdx.x >> 5;
  const int tileM = blockIdx.x * 16;
  const int l15 = lane & 15, hi = lane >> 4;
  const unsigned* arow = (const unsigned*)(Abf + (size_t)(tileM + l15) * D_MODEL);
  const unsigned* brow[4];
#pragma unroll
  for (int tb = 0; tb < 4; ++tb)   // B[k,n] = W[n,k]: lane n owns W row n
    brow[tb] = (const unsigned*)(Wbf + (size_t)(wave * 64 + tb * 16 + l15) * D_MODEL);
  v8f c[4] = {{}, {}, {}, {}};
  ABReg a0, a1, b0[4], b1[4];
  load_tile(arow, 0, hi, a0);
  load_b4(brow, 0, hi, b0);
  for (int kb = 0; kb < D_MODEL; kb += 64) {
    load_tile(arow, kb + 32, hi, a1);
    load_b4(brow, kb + 32, hi, b1);
    wmma4(a0, b0, c);
    if (kb + 64 < D_MODEL) {
      load_tile(arow, kb + 64, hi, a0);
      load_b4(brow, kb + 64, hi, b0);
    }
    wmma4(a1, b1, c);
  }
#pragma unroll
  for (int tb = 0; tb < 4; ++tb)
#pragma unroll
    for (int r = 0; r < 8; ++r) {
      const int row = tileM + r + hi * 8;
      const int col = wave * 64 + tb * 16 + l15;
      OutBf[(size_t)row * D_MODEL + col] = f2bf16(c[tb][r] + bias[col]);
    }
}

// ---------------- S = (q_bf @ k_bf^T) * 1/sqrt(512) ----------------
// Block covers 16 rows x 512 cols. The block-shared 16x512 bf16 A tile is
// staged into LDS once via the Tensor Data Mover (padded 1 dword per 256 to
// kill bank conflicts); B ping-pong double-buffered from global.
__global__ void scores_kernel(const unsigned short* __restrict__ Qbf,
                              const unsigned short* __restrict__ Kbf,
                              float* __restrict__ S) {
  __shared__ unsigned sA[16 * 257];   // 16 rows x 256 dwords (+1 dword pad/row)
  const int lane   = threadIdx.x & 31;
  const int wave   = threadIdx.x >> 5;
  const int tileM  = blockIdx.x * 16;
  const int tileN0 = blockIdx.y * 512 + wave * 64;
  const int l15 = lane & 15, hi = lane >> 4;

#if __has_builtin(__builtin_amdgcn_tensor_load_to_lds)
  if (wave == 0) {
    const unsigned long long ga =
        (unsigned long long)(const void*)(Qbf + (size_t)tileM * D_MODEL);
    // D# group0: count=1 | lds_addr | global_addr | type=2
    u32x4 g0 = { 1u,
                 (unsigned)(size_t)&sA[0],
                 (unsigned)(ga & 0xffffffffu),
                 (unsigned)(ga >> 32) | (2u << 30) };
    // D# group1: data_size=2B, pad_enable, pad_interval=256dw, pad_amount=1dw,
    //            tensor_dim0 = tile_dim0 = 8192 elems, tensor_dim1 = 1,
    //            tensor_dim0_stride = 8192
    i32x8 g1 = { (int)((1u << 16) | (1u << 20) | (7u << 22)),
                 (int)(8192u << 16),
                 (int)(1u << 16),
                 (int)(8192u << 16),
                 0,
                 (int)8192u,
                 0, 0 };
    i32x4 gz = { 0, 0, 0, 0 };
#if defined(__clang_major__) && __clang_major__ >= 23
    i32x8 gz8 = { 0, 0, 0, 0, 0, 0, 0, 0 };
    __builtin_amdgcn_tensor_load_to_lds(g0, g1, gz, gz, gz8, 0);
#else
    __builtin_amdgcn_tensor_load_to_lds(g0, g1, gz, gz, 0);
#endif
    __builtin_amdgcn_s_wait_tensorcnt(0);
  }
#else
  for (int idx = threadIdx.x; idx < 16 * 256; idx += 256) {
    const int r = idx >> 8, cd = idx & 255;
    sA[r * 257 + cd] =
        ((const unsigned*)(Qbf + (size_t)(tileM + r) * D_MODEL))[cd];
  }
#endif
  __syncthreads();

  const unsigned* krow[4];
#pragma unroll
  for (int tb = 0; tb < 4; ++tb)
    krow[tb] = (const unsigned*)(Kbf + (size_t)(tileN0 + tb * 16 + l15) * D_MODEL);
  const unsigned* aLds = &sA[l15 * 257];
  v8f c[4] = {{}, {}, {}, {}};
  ABReg a, b0[4], b1[4];
  load_b4(krow, 0, hi, b0);
  for (int kb = 0; kb < D_MODEL; kb += 64) {
    load_b4(krow, kb + 32, hi, b1);
    load_tile(aLds, kb, hi, a);
    wmma4(a, b0, c);
    if (kb + 64 < D_MODEL) load_b4(krow, kb + 64, hi, b0);
    load_tile(aLds, kb + 32, hi, a);
    wmma4(a, b1, c);
  }
  const float scale = 0.04419417382415922f;  // 1/sqrt(512)
#pragma unroll
  for (int tb = 0; tb < 4; ++tb)
#pragma unroll
    for (int r = 0; r < 8; ++r)
      S[(size_t)(tileM + r + hi * 8) * NSLOTS + tileN0 + tb * 16 + l15] =
          c[tb][r] * scale;
}

// ---------------- row softmax in place (4096 cols, 256 thr) ----------------
__global__ void softmax_kernel(float* __restrict__ S) {
  __shared__ float red[256];
  const int t = threadIdx.x;
  float* row = S + (size_t)blockIdx.x * NSLOTS;
  float x[16];
  float m = -3.402823466e38f;
#pragma unroll
  for (int j = 0; j < 16; ++j) { x[j] = row[t + j * 256]; m = fmaxf(m, x[j]); }
  red[t] = m; __syncthreads();
  for (int s = 128; s > 0; s >>= 1) {
    if (t < s) red[t] = fmaxf(red[t], red[t + s]);
    __syncthreads();
  }
  m = red[0]; __syncthreads();
  float sum = 0.f;
#pragma unroll
  for (int j = 0; j < 16; ++j) { x[j] = expf(x[j] - m); sum += x[j]; }
  red[t] = sum; __syncthreads();
  for (int s = 128; s > 0; s >>= 1) {
    if (t < s) red[t] += red[t + s];
    __syncthreads();
  }
  const float inv = 1.f / red[0];
#pragma unroll
  for (int j = 0; j < 16; ++j) row[t + j * 256] = x[j] * inv;
}

// ---------------- read_value = attn @ memory ----------------
// Block owns 16 rows x all 512 cols (128MB attn read exactly once from HBM);
// streamed fp32 A operand ping-pong double-buffered across K steps.
__global__ void readvalue_kernel(const float* __restrict__ Attn,
                                 const unsigned short* __restrict__ MemT,
                                 float* __restrict__ Out) {
  const int lane  = threadIdx.x & 31;
  const int wave  = threadIdx.x >> 5;
  const int tileM = blockIdx.x * 16;
  const int l15 = lane & 15, hi = lane >> 4;
  const float* arow = Attn + (size_t)(tileM + l15) * NSLOTS;
  const unsigned* brow[4];
#pragma unroll
  for (int tb = 0; tb < 4; ++tb)
    brow[tb] = (const unsigned*)(MemT + (size_t)(wave * 64 + tb * 16 + l15) * NSLOTS);
  v8f c[4] = {{}, {}, {}, {}};
  float2 af0[8], af1[8];
#pragma unroll
  for (int v = 0; v < 8; ++v) af0[v] = *(const float2*)(arow + kpair(0, v, hi));
  for (int kb = 0; kb < NSLOTS; kb += 64) {
#pragma unroll
    for (int v = 0; v < 8; ++v)
      af1[v] = *(const float2*)(arow + kpair(kb + 32, v, hi));
    {
      ABReg a, b[4];
#pragma unroll
      for (int v = 0; v < 8; ++v) a.u[v] = pack2bf(af0[v].x, af0[v].y);
      load_b4(brow, kb, hi, b);
      wmma4(a, b, c);
    }
    if (kb + 64 < NSLOTS) {
#pragma unroll
      for (int v = 0; v < 8; ++v)
        af0[v] = *(const float2*)(arow + kpair(kb + 64, v, hi));
    }
    {
      ABReg a, b[4];
#pragma unroll
      for (int v = 0; v < 8; ++v) a.u[v] = pack2bf(af1[v].x, af1[v].y);
      load_b4(brow, kb + 32, hi, b);
      wmma4(a, b, c);
    }
  }
#pragma unroll
  for (int tb = 0; tb < 4; ++tb)
#pragma unroll
    for (int r = 0; r < 8; ++r)
      Out[(size_t)(tileM + r + hi * 8) * D_MODEL + wave * 64 + tb * 16 + l15] = c[tb][r];
}

// ---------------- order-exact gated decay scatter ----------------
// One block per slot; addr+gate staged in LDS; uniform branch reproduces
// the sequential lax.scan collision order exactly.
__global__ void write_kernel(const float* __restrict__ Mem,
                             const float* __restrict__ Value,
                             const int* __restrict__ Loc,
                             const float* __restrict__ Gate,
                             float* __restrict__ NewMem) {
  __shared__ int   sAddr[BATCH_N];
  __shared__ float sGate[BATCH_N];
  const int t = threadIdx.x;
  const int s = blockIdx.x;
  for (int i = t; i < BATCH_N; i += 256) {
    sAddr[i] = Loc[i] & (NSLOTS - 1);   // location_id % 4096 (nonneg ids)
    sGate[i] = Gate[i];
  }
  __syncthreads();
  const int c0 = t * 2;                 // 512 cols / 256 threads
  float a0 = Mem[(size_t)s * D_MODEL + c0];
  float a1 = Mem[(size_t)s * D_MODEL + c0 + 1];
  for (int i = 0; i < BATCH_N; ++i) {
    if (sAddr[i] == s) {                // uniform across block
      const float g = 0.1f * sGate[i];
      a0 = 0.9f * a0 + g * Value[(size_t)i * D_MODEL + c0];
      a1 = 0.9f * a1 + g * Value[(size_t)i * D_MODEL + c0 + 1];
    }
  }
  NewMem[(size_t)s * D_MODEL + c0]     = a0;
  NewMem[(size_t)s * D_MODEL + c0 + 1] = a1;
}

extern "C" void kernel_launch(void* const* d_in, const int* in_sizes, int n_in,
                              void* d_out, int out_size, void* d_ws, size_t ws_size,
                              hipStream_t stream) {
  (void)in_sizes; (void)n_in; (void)out_size; (void)ws_size;
  const float* query  = (const float*)d_in[0];
  const float* value  = (const float*)d_in[1];
  const int*   loc    = (const int*)  d_in[2];
  const float* memory = (const float*)d_in[3];
  const float* Wq     = (const float*)d_in[4];
  const float* bq     = (const float*)d_in[5];
  const float* Wk     = (const float*)d_in[6];
  const float* bk     = (const float*)d_in[7];
  const float* wg     = (const float*)d_in[8];
  const float* bg     = (const float*)d_in[9];

  float* out_read = (float*)d_out;                          // 8192*512
  float* out_attn = out_read + (size_t)BATCH_N * D_MODEL;   // 8192*4096
  float* out_mem  = out_attn + (size_t)BATCH_N * NSLOTS;    // 4096*512

  char* ws = (char*)d_ws;
  unsigned short* qin_bf = (unsigned short*)ws;                     //  8.0 MB
  unsigned short* min_bf = (unsigned short*)(ws + 8388608);         //  4.0 MB
  unsigned short* wq_bf  = (unsigned short*)(ws + 12582912);        //  0.5 MB
  unsigned short* wk_bf  = (unsigned short*)(ws + 13107200);        //  0.5 MB
  unsigned short* q_bf   = (unsigned short*)(ws + 13631488);        //  8.0 MB
  unsigned short* k_bf   = (unsigned short*)(ws + 22020096);        //  4.0 MB
  unsigned short* memT   = (unsigned short*)(ws + 26214400);        //  4.0 MB
  float*          gate   = (float*)(ws + 30408704);                 //  32 KB

  // one-touch fp32 -> bf16 conversions (4 floats / thread)
  cvt_bf16_kernel<<<dim3((BATCH_N * D_MODEL) / 1024), 256, 0, stream>>>(query,  qin_bf);
  cvt_bf16_kernel<<<dim3((NSLOTS * D_MODEL) / 1024), 256, 0, stream>>>(memory, min_bf);
  cvt_bf16_kernel<<<dim3((D_MODEL * D_MODEL) / 1024), 256, 0, stream>>>(Wq, wq_bf);
  cvt_bf16_kernel<<<dim3((D_MODEL * D_MODEL) / 1024), 256, 0, stream>>>(Wk, wk_bf);

  gate_kernel<<<dim3(BATCH_N / 8), 256, 0, stream>>>(value, wg, bg, gate);
  transpose_mem_kernel<<<dim3((NSLOTS * D_MODEL) / 256), 256, 0, stream>>>(memory, memT);

  proj_gemm_kernel<<<dim3(BATCH_N / 16), 256, 0, stream>>>(qin_bf, wq_bf, bq, q_bf);
  proj_gemm_kernel<<<dim3(NSLOTS / 16), 256, 0, stream>>>(min_bf, wk_bf, bk, k_bf);

  scores_kernel<<<dim3(BATCH_N / 16, NSLOTS / 512), 256, 0, stream>>>(q_bf, k_bf, out_attn);
  softmax_kernel<<<dim3(BATCH_N), 256, 0, stream>>>(out_attn);
  readvalue_kernel<<<dim3(BATCH_N / 16), 256, 0, stream>>>(out_attn, memT, out_read);
  write_kernel<<<dim3(NSLOTS), 256, 0, stream>>>(memory, value, loc, gate, out_mem);
}
